// Model_53601191854858
// MI455X (gfx1250) — matrix operand order repaired and checked by reading, not compiled, not run on hardware
//
#include <hip/hip_runtime.h>
#include <hip/hip_bf16.h>

// ---------------- problem constants (from reference) ----------------
#define BN   32
#define TN   128
#define NN   512              // nodes
#define EN   4096             // edges
#define CN   4                // gcn channels
#define MN   (BN * TN)        // 4096 batched graph instances
#define K5   (NN * (1 + CN))  // 2560, projection K dim
#define ETOT (EN + NN)        // 4608 edges incl self loops

// ---------------- types for WMMA ----------------
typedef __bf16 bf16_t;
typedef __attribute__((ext_vector_type(8)))  __bf16 v8bf;
typedef __attribute__((ext_vector_type(16))) __bf16 v16bf;
typedef __attribute__((ext_vector_type(8)))  float  v8f;

// type expected by the async-to-LDS builtin (from hipcc diagnostic):
// int __attribute__((vector_size(16)))
typedef int v4i_t __attribute__((vector_size(4 * sizeof(int))));

// gfx1250 async global->LDS path (ASYNCcnt) with portable fallback
#if defined(__has_builtin)
#if __has_builtin(__builtin_amdgcn_global_load_async_to_lds_b128) && \
    __has_builtin(__builtin_amdgcn_s_wait_asynccnt)
#define USE_ASYNC_LDS 1
#endif
#endif

#ifdef USE_ASYNC_LDS
#define ASYNC_GPTR(p) ((__attribute__((address_space(1))) v4i_t*)(p))
#define ASYNC_LPTR(p) ((__attribute__((address_space(3))) v4i_t*)(p))
#endif

__device__ __forceinline__ unsigned short f2bf(float x) {
    unsigned int u = __float_as_uint(x);
    unsigned int r = u + 0x7FFFu + ((u >> 16) & 1u);   // round-to-nearest-even
    return (unsigned short)(r >> 16);
}

// ---------------- 1: init counts (self loop => start at 1) ----------------
__global__ void k_init_counts(int* __restrict__ counts) {
    int i = blockIdx.x * blockDim.x + threadIdx.x;
    if (i < NN) counts[i] = 1;
}

// ---------------- 2: histogram of dst ----------------
__global__ void k_count_edges(const int* __restrict__ edge_index, int* __restrict__ counts) {
    int e = blockIdx.x * blockDim.x + threadIdx.x;
    if (e < EN) atomicAdd(&counts[edge_index[EN + e]], 1);
}

// ---------------- 3: exclusive scan + dinv ----------------
__global__ void k_scan_dinv(const int* __restrict__ counts, int* __restrict__ offs,
                            float* __restrict__ dinv) {
    int tid = threadIdx.x;
    if (tid < NN) dinv[tid] = rsqrtf((float)counts[tid]);   // deg >= 1 always
    if (tid == 0) {
        int s = 0;
        for (int i = 0; i < NN; ++i) { offs[i] = s; s += counts[i]; }
        offs[NN] = s;
    }
}

// ---------------- 4: deterministic CSR build (tiny, serial) ----------------
__global__ void k_fill_csr(const int* __restrict__ edge_index, const int* __restrict__ offs,
                           const float* __restrict__ dinv, int* __restrict__ csr_src,
                           float* __restrict__ csr_w, int* __restrict__ cursor) {
    if (blockIdx.x != 0 || threadIdx.x != 0) return;
    for (int i = 0; i < NN; ++i) cursor[i] = 0;
    for (int e = 0; e < ETOT; ++e) {
        int s, d;
        if (e < EN) { s = edge_index[e]; d = edge_index[EN + e]; }
        else        { s = d = e - EN; }
        int pos = offs[d] + cursor[d]++;
        csr_src[pos] = s;
        csr_w[pos]   = dinv[s] * dinv[d];
    }
}

// ---------------- 5: GCN layer 1 (C_in = 1 -> scalar gather) ----------------
__global__ __launch_bounds__(256)
void k_gcn1(const float* __restrict__ x, const int* __restrict__ csr_src,
            const float* __restrict__ csr_w, const int* __restrict__ offs,
            const float* __restrict__ W1, const float* __restrict__ b1,
            float* __restrict__ h) {
    int idx = blockIdx.x * blockDim.x + threadIdx.x;   // M*N threads exactly
    int n = idx & (NN - 1);
    int m = idx >> 9;
    const float* xr = x + (size_t)m * NN;
    int p0 = offs[n], p1 = offs[n + 1];
    float g = 0.f;
    for (int p = p0; p < p1; ++p) g = fmaf(csr_w[p], xr[csr_src[p]], g);
    float4 hv;
    hv.x = fmaxf(fmaf(g, W1[0], b1[0]), 0.f);
    hv.y = fmaxf(fmaf(g, W1[1], b1[1]), 0.f);
    hv.z = fmaxf(fmaf(g, W1[2], b1[2]), 0.f);
    hv.w = fmaxf(fmaf(g, W1[3], b1[3]), 0.f);
    *(float4*)(h + (size_t)idx * 4) = hv;
}

// ------- 6: GCN layer 2 (gather then @W2), emit bf16 x_combined -------
__global__ __launch_bounds__(256)
void k_gcn2(const float* __restrict__ x, const float* __restrict__ h,
            const int* __restrict__ csr_src, const float* __restrict__ csr_w,
            const int* __restrict__ offs, const float* __restrict__ W2,
            const float* __restrict__ b2, unsigned short* __restrict__ xc) {
    int idx = blockIdx.x * blockDim.x + threadIdx.x;
    int n = idx & (NN - 1);
    int m = idx >> 9;
    const float* hb = h + (size_t)m * NN * 4;
    int p0 = offs[n], p1 = offs[n + 1];
    float a0 = 0.f, a1 = 0.f, a2 = 0.f, a3 = 0.f;
    for (int p = p0; p < p1; ++p) {
        float w = csr_w[p];
        float4 hv = *(const float4*)(hb + (size_t)csr_src[p] * 4);
        a0 = fmaf(w, hv.x, a0);
        a1 = fmaf(w, hv.y, a1);
        a2 = fmaf(w, hv.z, a2);
        a3 = fmaf(w, hv.w, a3);
    }
    unsigned short* o = xc + (size_t)m * K5 + n * (1 + CN);
    o[0] = f2bf(x[(size_t)m * NN + n]);
#pragma unroll
    for (int c = 0; c < CN; ++c) {
        float v = fmaf(a0, W2[0 * CN + c],
                  fmaf(a1, W2[1 * CN + c],
                  fmaf(a2, W2[2 * CN + c],
                  fmaf(a3, W2[3 * CN + c], b2[c]))));
        o[1 + c] = f2bf(v);
    }
}

// ---------------- 7: Wp f32 -> bf16 (row-major kept: already K-major) ----------------
__global__ void k_wp_bf16(const float* __restrict__ src, unsigned short* __restrict__ dst, int n) {
    int i = blockIdx.x * blockDim.x + threadIdx.x;
    if (i < n) dst[i] = f2bf(src[i]);
}

// ---------------- 8: projection GEMM via WMMA bf16 -> f32 ----------------
// out[m, j] = sum_k xc[m, k] * Wp[j, k] + bp[j]   (NT GEMM: both K-major)
//
// block = 256 threads = 8 waves (4 row-groups x 2 col-groups)
// block tile = 128 rows x 64 cols; wave tile = 32 rows x 32 cols (4 accumulators)
// B (Wp) slab [64 cols x 32 k] staged in LDS once per block, double-buffered,
// filled with GLOBAL_LOAD_ASYNC_TO_LDS_B128 (ASYNCcnt) when available.
__global__ __launch_bounds__(256)
void k_proj(const unsigned short* __restrict__ xcp, const unsigned short* __restrict__ wpp,
            const float* __restrict__ bp, float* __restrict__ out) {
    __shared__ __align__(16) unsigned short sB[2][64 * 32];   // 2 x 4 KB

    const int tid  = threadIdx.x;
    const int lane = tid & 31;
    const int wave = tid >> 5;
    const int rg = wave >> 1;          // row group 0..3
    const int cg = wave & 1;           // col group 0..1
    const int m0 = blockIdx.y * 128 + rg * 32;
    const int n0 = blockIdx.x * 64;
    const int lrow = lane & 15;        // A: row, B: col within 16-tile
    const int half = lane >> 4;        // lane half -> K sub-range
    const int aoff = half * 16;        // A layout: K run of 16 per lane half

    // cooperative B staging: each thread owns one 16B chunk per 32-k slab
    const int srow   = tid >> 2;            // 0..63 : column of Wp in block tile
    const int schunk = (tid & 3) * 8;       // element offset in slab
    const unsigned short* gB = wpp + (size_t)(n0 + srow) * K5 + schunk;
    unsigned short* lB[2] = { &sB[0][srow * 32 + schunk], &sB[1][srow * 32 + schunk] };

    const bf16_t* arow0 = (const bf16_t*)xcp + (size_t)(m0 + lrow) * K5;
    const bf16_t* arow1 = arow0 + (size_t)16 * K5;

    v8f acc[2][2] = {};

    // prologue: stage slab k=0 into buffer 0
#ifdef USE_ASYNC_LDS
    __builtin_amdgcn_global_load_async_to_lds_b128(ASYNC_GPTR(gB), ASYNC_LPTR(lB[0]), 0, 0);
    __builtin_amdgcn_s_wait_asynccnt(0);
#else
    *(uint4*)lB[0] = *(const uint4*)gB;
#endif
    __syncthreads();

    int buf = 0;
    for (int kk = 0; kk < K5; kk += 32) {
        // stage next slab into the other buffer (overlaps with WMMAs below)
        if (kk + 32 < K5) {
            const unsigned short* gnext = gB + kk + 32;
            unsigned short* lnext = lB[buf ^ 1];
#ifdef USE_ASYNC_LDS
            __builtin_amdgcn_global_load_async_to_lds_b128(ASYNC_GPTR(gnext),
                                                           ASYNC_LPTR(lnext), 0, 0);
#else
            *(uint4*)lnext = *(const uint4*)gnext;
#endif
        }

        // A fragments from global (L2-resident), contiguous K order k = 16*half + s (same as B)
        v8bf a0lo = *(const v8bf*)(arow0 + kk + aoff);
        v8bf a0hi = *(const v8bf*)(arow0 + kk + 8 + aoff);
        v8bf a1lo = *(const v8bf*)(arow1 + kk + aoff);
        v8bf a1hi = *(const v8bf*)(arow1 + kk + 8 + aoff);
        v16bf a0 = __builtin_shufflevector(a0lo, a0hi,
                                           0,1,2,3,4,5,6,7,8,9,10,11,12,13,14,15);
        v16bf a1 = __builtin_shufflevector(a1lo, a1hi,
                                           0,1,2,3,4,5,6,7,8,9,10,11,12,13,14,15);

        const unsigned short* lbuf = &sB[buf][0];
#pragma unroll
        for (int ct = 0; ct < 2; ++ct) {
            int rel = cg * 32 + ct * 16 + lrow;   // column within block tile
            // B layout: K run of 16 per lane-half -> one aligned 32B LDS read
            v16bf b = *(const v16bf*)(lbuf + rel * 32 + half * 16);
            acc[0][ct] = __builtin_amdgcn_wmma_f32_16x16x32_bf16(
                false, a0, false, b, (short)0, acc[0][ct], false, false);
            acc[1][ct] = __builtin_amdgcn_wmma_f32_16x16x32_bf16(
                false, a1, false, b, (short)0, acc[1][ct], false, false);
        }

#ifdef USE_ASYNC_LDS
        __builtin_amdgcn_s_wait_asynccnt(0);   // own async writes landed
#endif
        __syncthreads();                       // all waves done with buf + writes visible
        buf ^= 1;
    }

    // epilogue: C/D layout -> lane half h, VGPR j = row half*8+j, col = lane&15
#pragma unroll
    for (int ct = 0; ct < 2; ++ct) {
        int col = n0 + cg * 32 + ct * 16 + lrow;
        float bias = bp[col];
#pragma unroll
        for (int rt = 0; rt < 2; ++rt) {
            int r0 = m0 + rt * 16 + half * 8;
#pragma unroll
            for (int j = 0; j < 8; ++j)
                out[(size_t)(r0 + j) * NN + col] = acc[rt][ct][j] + bias;
        }
    }
}

// ---------------- host side ----------------
extern "C" void kernel_launch(void* const* d_in, const int* in_sizes, int n_in,
                              void* d_out, int out_size, void* d_ws, size_t ws_size,
                              hipStream_t stream) {
    const float* x_enc      = (const float*)d_in[0];
    const int*   edge_index = (const int*)d_in[4];
    const float* W1         = (const float*)d_in[5];
    const float* b1         = (const float*)d_in[6];
    const float* W2         = (const float*)d_in[7];
    const float* b2         = (const float*)d_in[8];
    const float* Wp         = (const float*)d_in[9];
    const float* bp         = (const float*)d_in[10];
    float* out = (float*)d_out;

    // workspace layout
    char* w = (char*)d_ws;
    size_t off = 0;
    int*   counts  = (int*)(w + off);   off += (size_t)NN * 4;
    int*   offs    = (int*)(w + off);   off += (size_t)(NN + 1) * 4;
    int*   cursor  = (int*)(w + off);   off += (size_t)NN * 4;
    float* dinv    = (float*)(w + off); off += (size_t)NN * 4;
    int*   csr_src = (int*)(w + off);   off += (size_t)ETOT * 4;
    float* csr_w   = (float*)(w + off); off += (size_t)ETOT * 4;
    off = (off + 255) & ~(size_t)255;
    float* h = (float*)(w + off);                     off += (size_t)MN * NN * 4 * 4;
    unsigned short* xc  = (unsigned short*)(w + off); off += (size_t)MN * K5 * 2;
    unsigned short* wpb = (unsigned short*)(w + off); off += (size_t)NN * K5 * 2;

    k_init_counts<<<(NN + 255) / 256, 256, 0, stream>>>(counts);
    k_count_edges<<<(EN + 255) / 256, 256, 0, stream>>>(edge_index, counts);
    k_scan_dinv<<<1, NN, 0, stream>>>(counts, offs, dinv);
    k_fill_csr<<<1, 1, 0, stream>>>(edge_index, offs, dinv, csr_src, csr_w, cursor);

    k_gcn1<<<(MN * NN) / 256, 256, 0, stream>>>(x_enc, csr_src, csr_w, offs, W1, b1, h);
    k_gcn2<<<(MN * NN) / 256, 256, 0, stream>>>(x_enc, h, csr_src, csr_w, offs, W2, b2, xc);

    k_wp_bf16<<<(NN * K5 + 255) / 256, 256, 0, stream>>>(Wp, wpb, NN * K5);

    dim3 grid(NN / 64, MN / 128);   // 8 x 32 = 256 blocks, 8 waves each
    k_proj<<<grid, 256, 0, stream>>>(xc, wpb, bp, out);
}
